// MixedEmbedding_30717606101108
// MI455X (gfx1250) — compile-verified, matrix operand order checked by reading, not run on hardware
//
#include <hip/hip_runtime.h>
#include <hip/hip_bf16.h>
#include <math.h>

// Problem constants (from reference)
#define K_DEG 5
#define L_DEG 4
#define M_F   32
#define D_IN  105          // (K+1)^2 + (L+1) + 2M = 36 + 5 + 64
#define D_HID 128
#define KP    136          // padded LDS row stride (bf16 elems): 272B = 68 dwords -> conflict-free
#define ROWS_PER_BLOCK 256
#define THREADS 256

typedef __attribute__((ext_vector_type(16))) __bf16 v16bf;
typedef __attribute__((ext_vector_type(8)))  float  v8f;

// float -> bf16 bits, round-to-nearest-even
__device__ __forceinline__ unsigned short f2bf(float x) {
  unsigned int u = __float_as_uint(x);
  u += 0x7FFFu + ((u >> 16) & 1u);
  return (unsigned short)(u >> 16);
}

union Frag { v16bf v; uint4 u[2]; };

// Load a 16-lane-striped bf16 WMMA fragment (A: row = m, B: row = n of W^T).
// Per the CDNA5 16-bit A layout, elements 0..7 are K = kbase + 8*half + 0..7,
// elements 8..15 are K = kbase + 16 + 8*half + 0..7 -> two contiguous 16B loads.
__device__ __forceinline__ v16bf load_frag(const unsigned short* __restrict__ row,
                                           int kbase, int half) {
  const unsigned short* p = row + kbase + (half << 3);
  Frag f;
  f.u[0] = *(const uint4*)(p);
  f.u[1] = *(const uint4*)(p + 16);
  return f.v;
}

__device__ __forceinline__ v8f wmma_bf16(v16bf a, v16bf b, v8f c) {
  // (neg_a, A, neg_b, B, c_mod, C, reuse_a, reuse_b)
  return __builtin_amdgcn_wmma_f32_16x16x32_bf16(false, a, false, b, (short)0, c,
                                                 false, false);
}

// One MLP layer for this wave's 32 rows: D = A(32x128) * W^T + bias, optional relu.
// Output goes to LDS (bf16) and/or global fp32.
__device__ __forceinline__ void gemm_layer(
    const unsigned short* __restrict__ sA,    // [256][KP] activations (bf16)
    const unsigned short* __restrict__ sW,    // [128][KP] weights^T (bf16)
    const float* __restrict__ bias,           // [128] global fp32
    unsigned short* __restrict__ sOut,        // [256][KP] or nullptr
    float* __restrict__ gOut,                 // global out or nullptr
    int blockBase, int B, int wave, int half, int ln, bool relu)
{
  const int rowBase = wave * 32;

  v8f acc[2][8];
  const v8f vzero = {0.f, 0.f, 0.f, 0.f, 0.f, 0.f, 0.f, 0.f};
#pragma unroll
  for (int mt = 0; mt < 2; ++mt)
#pragma unroll
    for (int nt = 0; nt < 8; ++nt) acc[mt][nt] = vzero;

#pragma unroll
  for (int ks = 0; ks < 4; ++ks) {
    const int kb = ks * 32;
    v16bf a0 = load_frag(sA + (rowBase +      ln) * KP, kb, half);
    v16bf a1 = load_frag(sA + (rowBase + 16 + ln) * KP, kb, half);
#pragma unroll
    for (int nt = 0; nt < 8; ++nt) {
      v16bf b = load_frag(sW + (nt * 16 + ln) * KP, kb, half);
      acc[0][nt] = wmma_bf16(a0, b, acc[0][nt]);
      acc[1][nt] = wmma_bf16(a1, b, acc[1][nt]);
    }
  }

  // Epilogue: C/D layout -> lane l holds col n = 16*nt + ln, rows r + 8*half.
#pragma unroll
  for (int nt = 0; nt < 8; ++nt) {
    const int n = nt * 16 + ln;
    const float bn = bias[n];
#pragma unroll
    for (int mt = 0; mt < 2; ++mt) {
#pragma unroll
      for (int r = 0; r < 8; ++r) {
        const int rl = rowBase + mt * 16 + r + half * 8;  // row within block
        float v = acc[mt][nt][r] + bn;
        if (relu) v = fmaxf(v, 0.f);
        if (sOut) sOut[rl * KP + n] = f2bf(v);
        if (gOut) {
          const int rg = blockBase + rl;
          if (rg < B) gOut[(size_t)rg * D_HID + n] = v;
        }
      }
    }
  }
}

__global__ __launch_bounds__(THREADS, 1)
void MixedEmbedding_30717606101108_kernel(
    const float* __restrict__ loc, const float* __restrict__ fw,
    const float* __restrict__ fb,
    const float* __restrict__ w1, const float* __restrict__ b1,
    const float* __restrict__ w2, const float* __restrict__ b2,
    const float* __restrict__ w3, const float* __restrict__ b3,
    float* __restrict__ out, int B)
{
  __shared__ __align__(16) unsigned short sW1[D_HID * KP];
  __shared__ __align__(16) unsigned short sW2[D_HID * KP];
  __shared__ __align__(16) unsigned short sW3[D_HID * KP];
  __shared__ __align__(16) unsigned short sF [ROWS_PER_BLOCK * KP];
  __shared__ __align__(16) unsigned short sH [ROWS_PER_BLOCK * KP];

  const int tid  = threadIdx.x;
  const int wave = tid >> 5;
  const int lane = tid & 31;
  const int half = lane >> 4;
  const int ln   = lane & 15;
  const int blockBase = blockIdx.x * ROWS_PER_BLOCK;

  // ---- Stage weights: transpose to [n][k], bf16, zero-pad k in [105,128) ----
  for (int idx = tid; idx < D_HID * D_HID; idx += THREADS) {
    const int k = idx >> 7;      // consecutive threads -> consecutive n (coalesced)
    const int n = idx & 127;
    sW1[n * KP + k] = (k < D_IN) ? f2bf(w1[k * D_HID + n]) : (unsigned short)0;
    sW2[n * KP + k] = f2bf(w2[k * D_HID + n]);
    sW3[n * KP + k] = f2bf(w3[k * D_HID + n]);
  }

  // ---- Per-thread feature construction (fp32 VALU, one row per thread) ----
  {
    const int row = blockBase + tid;
    const int rc  = row < B ? row : (B - 1);
    const float* p = loc + (size_t)rc * 6;
    const float x = p[0], y = p[1], zc = p[2];
    const float hx = x - p[3], hy = y - p[4], hz = zc - p[5];
    const float theta = atan2f(hy, hx);
    const float phi   = atan2f(hz, sqrtf(hx * hx + hy * hy));

    float Vx[K_DEG + 1], Vy[K_DEG + 1], Vz[L_DEG + 1];
    Vx[0] = 1.f; Vx[1] = x;
    Vy[0] = 1.f; Vy[1] = y;
#pragma unroll
    for (int n = 1; n < K_DEG; ++n) {
      Vx[n + 1] = ((2 * n + 1) * x * Vx[n] - n * Vx[n - 1]) / (float)(n + 1);
      Vy[n + 1] = ((2 * n + 1) * y * Vy[n] - n * Vy[n - 1]) / (float)(n + 1);
    }
    Vz[0] = 1.f; Vz[1] = zc;
#pragma unroll
    for (int n = 1; n < L_DEG; ++n)
      Vz[n + 1] = ((2 * n + 1) * zc * Vz[n] - n * Vz[n - 1]) / (float)(n + 1);

    unsigned short* fr = sF + tid * KP;
#pragma unroll
    for (int i = 0; i <= K_DEG; ++i)
#pragma unroll
      for (int j = 0; j <= K_DEG; ++j)
        fr[i * (K_DEG + 1) + j] = f2bf(Vx[i] * Vy[j]);
#pragma unroll
    for (int i = 0; i <= L_DEG; ++i) fr[36 + i] = f2bf(Vz[i]);

    const float s = 0.125f;  // 1/sqrt(2M) = 1/8
#pragma unroll 4
    for (int m = 0; m < M_F; ++m) {
      const float am = theta * fw[2 * m] + phi * fw[2 * m + 1] + fb[m];
      fr[41 + m]       = f2bf(__cosf(am) * s);
      fr[41 + M_F + m] = f2bf(__sinf(am) * s);
    }
#pragma unroll
    for (int k2 = D_IN; k2 < D_HID; ++k2) fr[k2] = 0;
  }

  __syncthreads();  // weights + features visible; layers below are wave-private

  gemm_layer(sF, sW1, b1, sH, nullptr, blockBase, B, wave, half, ln, true);
  gemm_layer(sH, sW2, b2, sF, nullptr, blockBase, B, wave, half, ln, true);
  gemm_layer(sF, sW3, b3, nullptr, out,  blockBase, B, wave, half, ln, false);
}

extern "C" void kernel_launch(void* const* d_in, const int* in_sizes, int n_in,
                              void* d_out, int out_size, void* d_ws, size_t ws_size,
                              hipStream_t stream) {
  (void)n_in; (void)out_size; (void)d_ws; (void)ws_size;
  const float* loc = (const float*)d_in[0];
  const float* fw  = (const float*)d_in[1];
  const float* fb  = (const float*)d_in[2];
  const float* w1  = (const float*)d_in[3];
  const float* b1  = (const float*)d_in[4];
  const float* w2  = (const float*)d_in[5];
  const float* b2  = (const float*)d_in[6];
  const float* w3  = (const float*)d_in[7];
  const float* b3  = (const float*)d_in[8];
  float* out = (float*)d_out;

  const int B = in_sizes[0] / 6;
  const int grid = (B + ROWS_PER_BLOCK - 1) / ROWS_PER_BLOCK;
  hipLaunchKernelGGL(MixedEmbedding_30717606101108_kernel,
                     dim3(grid), dim3(THREADS), 0, stream,
                     loc, fw, fb, w1, b1, w2, b2, w3, b3, out, B);
}
